// MIMOcom_26156350833151
// MI455X (gfx1250) — matrix operand order repaired and checked by reading, not compiled, not run on hardware
//
#include <hip/hip_runtime.h>

// Problem constants (from reference): M=512, N=64, C=128, H=W=8 -> P=Q=64
#define MM   512
#define NN   64
#define CC   128
#define PP   64   // H*W
#define EPSF 0.001f

// LDS row pitch in bf16 elements: 128 data + 8 pad = 272 B per row.
// 272 B = 68 dwords == 4 (mod 64 banks): the 16 fragment columns start at
// banks 0,4,...,60 and their 16B reads tile all 64 banks conflict-free.
#define LDSROW 136

typedef __attribute__((ext_vector_type(16))) __bf16 v16bf;
typedef __attribute__((ext_vector_type(8)))  __bf16 v8bf;
typedef __attribute__((ext_vector_type(8)))  float  v8f;
typedef int v4i __attribute__((vector_size(16)));

union V16 { v16bf v; v8bf h[2]; };

static __device__ __forceinline__ __bf16 f2bf(float x) { return (__bf16)x; }

#if defined(__has_builtin)
#if __has_builtin(__builtin_amdgcn_global_load_async_to_lds_b128)
#define HAVE_ASYNC_LDS 1
#endif
#endif

#ifdef HAVE_ASYNC_LDS
#define AS1 __attribute__((address_space(1)))
#define AS3 __attribute__((address_space(3)))
static __device__ __forceinline__ AS1 v4i* to_global(const void* p) {
    return (AS1 v4i*)(unsigned long long)(uintptr_t)p;
}
static __device__ __forceinline__ AS3 v4i* to_shared(void* p) {
    // low 32 bits of a generic LDS pointer are the LDS byte offset
    return (AS3 v4i*)(unsigned int)(uintptr_t)p;
}
#endif

// ---------------------------------------------------------------------------
// Kernel 1: split-transpose A [M, C, P] -> Ah/Al [M*P, C] bf16 hi/lo planes
// row r = m*64 + p, K-contiguous so WMMA A-fragments are contiguous loads.
// ---------------------------------------------------------------------------
__global__ void split_a_kernel(const float* __restrict__ A,
                               __bf16* __restrict__ Ah,
                               __bf16* __restrict__ Al) {
    int idx = blockIdx.x * blockDim.x + threadIdx.x;   // 0 .. M*P*C-1
    int k = idx & (CC - 1);
    int r = idx >> 7;            // r = m*64 + p
    int m = r >> 6;
    int p = r & 63;
    float x = A[((size_t)m * CC + k) * PP + p];
    __bf16 hi = f2bf(x);
    Ah[idx] = hi;
    Al[idx] = f2bf(x - (float)hi);
}

// ---------------------------------------------------------------------------
// Kernel 2: G[n,q,d] = sum_c B[n,c,q] * Wg[c,d]  (0.13 GFLOP, scalar is fine)
// stored as bf16 hi/lo planes [N*Q, C], K(=d)-contiguous for B-fragments.
// ---------------------------------------------------------------------------
__global__ void g_kernel(const float* __restrict__ B,
                         const float* __restrict__ Wg,
                         __bf16* __restrict__ Gh,
                         __bf16* __restrict__ Gl) {
    int idx = blockIdx.x * blockDim.x + threadIdx.x;   // 0 .. N*Q*C-1
    int d = idx & (CC - 1);
    int q = (idx >> 7) & 63;
    int n = idx >> 13;
    const float* Bn = B + (size_t)n * CC * PP;
    float s = 0.0f;
#pragma unroll 8
    for (int c = 0; c < CC; ++c)
        s += Bn[c * PP + q] * Wg[c * CC + d];
    __bf16 hi = f2bf(s);
    Gh[idx] = hi;
    Gl[idx] = f2bf(s - (float)hi);
}

// ---------------------------------------------------------------------------
// Main kernel: block = 8 waves share one n; G[n] staged in LDS (async copy).
// Each wave: one 16-row (m,p) tile; 4 q-tiles x 4 K-chunks x 3 bf16x3 WMMAs,
// fused max-over-q + relu + eps epilogue.
// ---------------------------------------------------------------------------
__global__ __launch_bounds__(256) void
mimo_wmma_kernel(const __bf16* __restrict__ Ah, const __bf16* __restrict__ Al,
                 const __bf16* __restrict__ Gh, const __bf16* __restrict__ Gl,
                 float* __restrict__ out) {
    __shared__ __bf16 sGh[PP * LDSROW];   // 17408 B
    __shared__ __bf16 sGl[PP * LDSROW];   // 17408 B

    const int lane = threadIdx.x & 31;
    const int wave = threadIdx.x >> 5;
    const int lm   = lane & 15;       // fragment column / A-row within tile
    const int hh   = lane >> 4;       // half-wave select

    const int rt = blockIdx.x * 8 + wave;   // 16-row tile id, 0..2047
    const int m  = rt >> 2;                 // 4 row-tiles per m (P=64)
    const int p0 = (rt & 3) << 4;
    const int n  = blockIdx.y;

    // ---- preload A fragments (hi & lo) for all 4 K-chunks (registers) ----
    // ISA 16-bit A 16x32 layout: per lane, elems 0-7 = K 8h..8h+7,
    // elems 8-15 = K 16+8h..16+8h+7.  Row = m*64 + p0 + lm (K-contiguous).
    const __bf16* arh = Ah + (size_t)(m * PP + p0 + lm) * CC;
    const __bf16* arl = Al + (size_t)(m * PP + p0 + lm) * CC;
    V16 afh[4], afl[4];
#pragma unroll
    for (int c = 0; c < 4; ++c) {
        const int kb = c * 32;
        afh[c].h[0] = *(const v8bf*)(arh + kb + 8 * hh);
        afh[c].h[1] = *(const v8bf*)(arh + kb + 16 + 8 * hh);
        afl[c].h[0] = *(const v8bf*)(arl + kb + 8 * hh);
        afl[c].h[1] = *(const v8bf*)(arl + kb + 16 + 8 * hh);
    }

    // ---- stage Gh[n], Gl[n] (32 KB) into padded LDS, once per block ------
    {
        const uint4* gh4 = (const uint4*)(Gh + (size_t)n * PP * CC); // 16B chunks
        const uint4* gl4 = (const uint4*)(Gl + (size_t)n * PP * CC);
#pragma unroll
        for (int t = 0; t < 4; ++t) {
            int i = t * 256 + threadIdx.x;        // 0..1023 (16B chunks)
            int q = i >> 4;
            int j = i & 15;
            uint4* dh = (uint4*)(sGh + q * LDSROW) + j;
            uint4* dl = (uint4*)(sGl + q * LDSROW) + j;
#ifdef HAVE_ASYNC_LDS
            __builtin_amdgcn_global_load_async_to_lds_b128(
                to_global(gh4 + q * 16 + j), to_shared(dh), 0, 0);
            __builtin_amdgcn_global_load_async_to_lds_b128(
                to_global(gl4 + q * 16 + j), to_shared(dl), 0, 0);
#else
            *dh = gh4[q * 16 + j];
            *dl = gl4[q * 16 + j];
#endif
        }
#ifdef HAVE_ASYNC_LDS
        asm volatile("s_wait_asynccnt 0" ::: "memory");
#endif
        __syncthreads();
    }

    // ---- GEMM + running max over the 4 q-tiles ---------------------------
    v8f vmax;
#pragma unroll
    for (int qt = 0; qt < 4; ++qt) {
        // ISA 16-bit B 32x16 layout: per lane (col N=lm), elems 0-7 = K 8h..8h+7, elems 8-15 = K 16+8h..16+8h+7
        const __bf16* sgh = sGh + (qt * 16 + lm) * LDSROW + 8 * hh;
        const __bf16* sgl = sGl + (qt * 16 + lm) * LDSROW + 8 * hh;

        v8f acc = {0.f, 0.f, 0.f, 0.f, 0.f, 0.f, 0.f, 0.f};
#pragma unroll
        for (int c = 0; c < 4; ++c) {
            const int kb = c * 32;
            V16 bh, bl;
            bh.h[0] = *(const v8bf*)(sgh + kb);
            bh.h[1] = *(const v8bf*)(sgh + kb + 16);
            bl.h[0] = *(const v8bf*)(sgl + kb);
            bl.h[1] = *(const v8bf*)(sgl + kb + 16);
            // bf16x3: hi*hi + hi*lo + lo*hi  (fp32 accumulate)
            acc = __builtin_amdgcn_wmma_f32_16x16x32_bf16(
                      false, afh[c].v, false, bh.v, (short)0, acc, false, false);
            acc = __builtin_amdgcn_wmma_f32_16x16x32_bf16(
                      false, afh[c].v, false, bl.v, (short)0, acc, false, false);
            acc = __builtin_amdgcn_wmma_f32_16x16x32_bf16(
                      false, afl[c].v, false, bh.v, (short)0, acc, false, false);
        }
        if (qt == 0) {
            vmax = acc;
        } else {
#pragma unroll
            for (int v = 0; v < 8; ++v) vmax[v] = fmaxf(vmax[v], acc[v]);
        }
    }

    // ---- horizontal max over the 16 q-columns (lanes within each half) ---
#pragma unroll
    for (int v = 0; v < 8; ++v) {
        float x = vmax[v];
#pragma unroll
        for (int off = 8; off >= 1; off >>= 1)
            x = fmaxf(x, __shfl_xor(x, off, 32));
        vmax[v] = x;
    }

    // ---- epilogue: relu + eps, 16 predicated stores per wave -------------
    // D layout: VGPR v holds M=v (lanes 0-15) / M=v+8 (lanes 16-31);
    // after reduction lanes lm==0 of each half hold the row maxima.
    if (lm == 0) {
#pragma unroll
        for (int v = 0; v < 8; ++v) {
            int p = p0 + v + 8 * hh;
            out[(size_t)m * (NN * PP) + n * PP + p] =
                fmaxf(vmax[v], 0.0f) + EPSF;
        }
    }
}

// ---------------------------------------------------------------------------
extern "C" void kernel_launch(void* const* d_in, const int* in_sizes, int n_in,
                              void* d_out, int out_size, void* d_ws, size_t ws_size,
                              hipStream_t stream) {
    const float* A  = (const float*)d_in[0];   // [512,128,8,8]
    const float* B  = (const float*)d_in[1];   // [64,128,8,8]
    const float* Wg = (const float*)d_in[2];   // [128,128]
    float* out = (float*)d_out;                // [512,64,64]

    // workspace layout (bf16 planes): Ah | Al | Gh | Gl  (~18.9 MB total)
    const size_t a_elems = (size_t)MM * PP * CC;   // 4,194,304
    const size_t g_elems = (size_t)NN * PP * CC;   //   524,288
    __bf16* Ah = (__bf16*)d_ws;
    __bf16* Al = Ah + a_elems;
    __bf16* Gh = Al + a_elems;
    __bf16* Gl = Gh + g_elems;

    split_a_kernel<<<(int)(a_elems / 256), 256, 0, stream>>>(A, Ah, Al);
    g_kernel<<<(int)(g_elems / 256), 256, 0, stream>>>(B, Wg, Gh, Gl);

    // 2048 row-tiles / 8 waves per block = 256 blocks in x; n in y.
    dim3 grid(256, NN);
    mimo_wmma_kernel<<<grid, 256, 0, stream>>>(Ah, Al, Gh, Gl, out);
}
